// EdgeBiasedAttention_12438225290017
// MI455X (gfx1250) — matrix operand order repaired and checked by reading, not compiled, not run on hardware
//
#include <hip/hip_runtime.h>
#include <hip/hip_bf16.h>

// EdgeBiasedAttention for MI455X (gfx1250, wave32, WMMA + async global->LDS DMA).
// [1] QKV projection (f32 -> f16, q pre-scaled by 1/sqrt(HD))
// [2] flash-style edge-biased attention per (b, head):
//     - K tiles double-buffered in LDS via global_load_async_to_lds_b128 (block-shared)
//     - edge/adj tiles double-buffered in per-wave LDS via async DMA (no barrier needed)
//     - scores / P.V on v_wmma_f32_16x16x32_f16, online softmax
// [3] output projection (f16 x f16 -> f32)

#define B_  32
#define N_  512
#define D_  256
#define H_  8
#define HD_ 32

typedef __attribute__((ext_vector_type(16))) _Float16 v16h;
typedef __attribute__((ext_vector_type(8)))  float    v8f;

#if defined(__gfx1250__)
#define USE_ASYNC 1
#else
#define USE_ASYNC 0
#endif

#if USE_ASYNC
// Async DMA: LDS[l..l+15] = MEM[g..g+15], tracked by ASYNCcnt (per-lane addresses).
// Inline asm is the toolchain-portable route (builtin pointer AS typing differs).
static __device__ __forceinline__ void async_ld16(const void* g, void* l) {
  const unsigned           lds32 = (unsigned)(unsigned long long)l;  // LDS = flat[31:0]
  const unsigned long long ga    = (unsigned long long)g;
  asm volatile("global_load_async_to_lds_b128 %0, %1, off"
               :: "v"(lds32), "v"(ga) : "memory");
}
static __device__ __forceinline__ void wait_async0() {
  asm volatile("s_wait_asynccnt 0x0" ::: "memory");
}
#endif

// ---- fragment loaders ------------------------------------------------------
// A-matrix 16x32 f16: lane holds row (lane&15);
// half t -> K = kb16+t, kb16 = 16*(lane>=16) (contiguous, matches B frags).
static __device__ __forceinline__ v16h afrag_f32(const float* p) {
  v16h f;
#pragma unroll
  for (int j = 0; j < 16; ++j) f[j] = (_Float16)p[j];
  return f;
}
static __device__ __forceinline__ v16h afrag_f16(const _Float16* p) {
  v16h f;
#pragma unroll
  for (int j = 0; j < 16; ++j) f[j] = p[j];
  return f;
}
// B-matrix 32x16 f16: lane holds column (lane&15); half t -> K = kb16+t,
// kb16 = 16*(lane>=16) -> contiguous 16-element read for x @ W^T style GEMMs.
static __device__ __forceinline__ v16h bfrag_f32(const float* p) {
  v16h f;
#pragma unroll
  for (int t = 0; t < 16; ++t) f[t] = (_Float16)p[t];
  return f;
}
static __device__ __forceinline__ v16h bfrag_f16(const _Float16* p) {
  v16h f;
#pragma unroll
  for (int t = 0; t < 16; ++t) f[t] = p[t];
  return f;
}
static __device__ __forceinline__ v8f wmma16(v16h a, v16h b, v8f c) {
  return __builtin_amdgcn_wmma_f32_16x16x32_f16(false, a, false, b, (short)0, c, false, false);
}

// ---- [1] QKV projection ----------------------------------------------------
__global__ __launch_bounds__(256) void qkv_proj(
    const float* __restrict__ h,
    const float* __restrict__ Wq, const float* __restrict__ Wk, const float* __restrict__ Wv,
    _Float16* __restrict__ qo, _Float16* __restrict__ ko, _Float16* __restrict__ vo) {
  const int lane = threadIdx.x & 31;
  const int wid  = blockIdx.x * 8 + (threadIdx.x >> 5);  // 0..3071
  const int mat  = wid % 3;
  const int m0   = (wid / 3) * 16;
  const float* W     = (mat == 0) ? Wq : (mat == 1 ? Wk : Wv);
  _Float16*    dst   = (mat == 0) ? qo : (mat == 1 ? ko : vo);
  const float  scale = (mat == 0) ? 0.17677669529663687f : 1.0f;  // 1/sqrt(32)

  const int r    = lane & 15;
  const int kb8  = (lane >> 4) << 3;
  const int kb16 = (lane >> 4) << 4;
  const int mh   = (lane >> 4) << 3;

  v16h a[8];
#pragma unroll
  for (int kk = 0; kk < 8; ++kk)
    a[kk] = afrag_f32(h + (size_t)(m0 + r) * D_ + kk * 32 + kb16);

  for (int nt = 0; nt < 16; ++nt) {
    v8f acc = {0.f, 0.f, 0.f, 0.f, 0.f, 0.f, 0.f, 0.f};
#pragma unroll
    for (int kk = 0; kk < 8; ++kk) {
      v16h b = bfrag_f32(W + (size_t)(nt * 16 + r) * D_ + kk * 32 + kb16);
      acc = wmma16(a[kk], b, acc);
    }
#pragma unroll
    for (int rr = 0; rr < 8; ++rr)
      dst[(size_t)(m0 + rr + mh) * D_ + nt * 16 + r] = (_Float16)(acc[rr] * scale);
  }
}

// ---- [2] flash attention with edge bias + adjacency mask -------------------
__global__ __launch_bounds__(256) void edge_attn(
    const _Float16* __restrict__ q, const _Float16* __restrict__ k,
    const _Float16* __restrict__ v, const float* __restrict__ edge,
    const unsigned char* __restrict__ adj, const float* __restrict__ We,
    _Float16* __restrict__ att) {
  __shared__ _Float16 vT[HD_][N_ + 8];                                   // ~33 KB
  __shared__ _Float16 Pbuf[8][16 * 32];                                  //   8 KB
  __shared__ __attribute__((aligned(16))) _Float16 Kbuf[2][32][40];      //   5 KB
  __shared__ __attribute__((aligned(16))) float    Ebuf[8][2][16 * 32];  //  32 KB
  __shared__ __attribute__((aligned(16))) unsigned char Abuf[8][2][16 * 32];  // 8 KB

  const int b    = blockIdx.x / H_;
  const int hh   = blockIdx.x % H_;
  const int lane = threadIdx.x & 31;
  const int wave = threadIdx.x >> 5;
  const int r    = lane & 15;
  const int kb8  = (lane >> 4) << 3;
  const int kb16 = (lane >> 4) << 4;
  const int mh   = (lane >> 4) << 3;

  const size_t head_base = (size_t)b * N_ * D_ + (size_t)hh * HD_;
  const _Float16* kg = k + head_base;

  // stage V^T once per block (read coalesced, write transposed)
  for (int idx = threadIdx.x; idx < N_ * HD_; idx += 256) {
    const int j = idx >> 5, d = idx & 31;
    vT[d][j] = v[head_base + (size_t)j * D_ + d];
  }
  __syncthreads();

  const float  we    = We[hh];
  const size_t ebase = (size_t)b * N_ * N_;

  // stage one 32x32 K tile (rows j0..j0+31, this head's 32 dims) into Kbuf[buf]
  auto stage_k = [&](int buf, int j0) {
#if USE_ASYNC
    if (wave == 0) {
      const _Float16* g = kg + (size_t)(j0 + lane) * D_;   // 64B per row
      _Float16*       l = &Kbuf[buf][lane][0];
#pragma unroll
      for (int c = 0; c < 4; ++c) async_ld16(g + c * 8, l + c * 8);
    }
#else
    for (int idx = threadIdx.x; idx < 32 * HD_; idx += 256) {
      const int jl = idx >> 5, d = idx & 31;
      Kbuf[buf][jl][d] = kg[(size_t)(j0 + jl) * D_ + d];
    }
#endif
  };

  // stage this wave's 16x32 edge + adj tiles (rows m0.., cols j0..) per-wave
  auto stage_e = [&](int buf, int m0, int j0) {
    const float*         eg = edge + ebase + (size_t)m0 * N_ + j0;
    const unsigned char* ag = adj  + ebase + (size_t)m0 * N_ + j0;
#if USE_ASYNC
    {
      const int row = lane >> 3, ch = lane & 7;  // 4 rows x 8 chunks per instr
#pragma unroll
      for (int c = 0; c < 4; ++c)
        async_ld16(eg + (size_t)(c * 4 + row) * N_ + ch * 4,
                   &Ebuf[wave][buf][(c * 4 + row) * 32 + ch * 4]);
      const int arow = lane >> 1, ach = lane & 1;  // 16 rows x 2 chunks, one instr
      async_ld16(ag + (size_t)arow * N_ + ach * 16,
                 &Abuf[wave][buf][arow * 32 + ach * 16]);
    }
#else
    for (int idx = lane; idx < 16 * 32; idx += 32) {
      const int rr = idx >> 5, cc = idx & 31;
      Ebuf[wave][buf][idx] = eg[(size_t)rr * N_ + cc];
      Abuf[wave][buf][idx] = ag[(size_t)rr * N_ + cc];
    }
#endif
  };

  for (int pass = 0; pass < 4; ++pass) {
    const int  m0 = (pass * 8 + wave) * 16;
    const v16h qa = afrag_f16(q + head_base + (size_t)(m0 + r) * D_ + kb16);

    v8f   acc0 = {0.f,0.f,0.f,0.f,0.f,0.f,0.f,0.f};
    v8f   acc1 = {0.f,0.f,0.f,0.f,0.f,0.f,0.f,0.f};
    float rmax[8], rsum[8];
#pragma unroll
    for (int t = 0; t < 8; ++t) { rmax[t] = -3.0e38f; rsum[t] = 0.f; }

    stage_k(0, 0);
    stage_e(0, m0, 0);

    for (int jt = 0; jt < N_ / 32; ++jt) {
      const int cur = jt & 1;
      const int j0  = jt * 32;
#if USE_ASYNC
      wait_async0();                        // own edge DMA (+K for wave 0) done
#endif
      __syncthreads();                      // K tile visible block-wide
      if (jt + 1 < N_ / 32) {
        stage_k(cur ^ 1, j0 + 32);          // overlap next tile with compute
        stage_e(cur ^ 1, m0, j0 + 32);
      }
      if (jt + 4 < N_ / 32)                 // warm L2 ahead of the DMA stream
        __builtin_prefetch(edge + ebase + (size_t)m0 * N_ + j0 + 128 + lane, 0, 0);

      // scores: Q(16x32) x K^T -> two 16x16 tiles (HD=32 == one WMMA K step)
      const v16h kf0 = bfrag_f16(&Kbuf[cur][r][kb16]);
      const v16h kf1 = bfrag_f16(&Kbuf[cur][16 + r][kb16]);
      const v8f  z   = {0.f,0.f,0.f,0.f,0.f,0.f,0.f,0.f};
      v8f s0 = wmma16(qa, kf0, z);
      v8f s1 = wmma16(qa, kf1, z);

#pragma unroll
      for (int t = 0; t < 8; ++t) {
        const int erow = (t + mh) * 32;
        float x0 = s0[t] + we * Ebuf[wave][cur][erow + r];
        float x1 = s1[t] + we * Ebuf[wave][cur][erow + 16 + r];
        if (!Abuf[wave][cur][erow + r])      x0 = -1.0e30f;
        if (!Abuf[wave][cur][erow + 16 + r]) x1 = -1.0e30f;

        // row max over 32 keys (16-lane butterfly, wave32-safe)
        float tm = fmaxf(x0, x1);
        tm = fmaxf(tm, __shfl_xor(tm, 1));
        tm = fmaxf(tm, __shfl_xor(tm, 2));
        tm = fmaxf(tm, __shfl_xor(tm, 4));
        tm = fmaxf(tm, __shfl_xor(tm, 8));
        const float mnew = fmaxf(rmax[t], tm);
        const float corr = __expf(rmax[t] - mnew);
        const float p0   = __expf(x0 - mnew);
        const float p1   = __expf(x1 - mnew);
        float ls = p0 + p1;
        ls += __shfl_xor(ls, 1);
        ls += __shfl_xor(ls, 2);
        ls += __shfl_xor(ls, 4);
        ls += __shfl_xor(ls, 8);
        rsum[t] = rsum[t] * corr + ls;
        rmax[t] = mnew;
        acc0[t] *= corr;
        acc1[t] *= corr;
        Pbuf[wave][erow + r]      = (_Float16)p0;
        Pbuf[wave][erow + 16 + r] = (_Float16)p1;
      }

      // O += P(16x32) x V(32x32): A-frag from LDS Pbuf, B-frags from LDS V^T
      const v16h pa  = afrag_f16(&Pbuf[wave][r * 32 + kb16]);
      const v16h vb0 = bfrag_f16(&vT[r][j0 + kb16]);
      const v16h vb1 = bfrag_f16(&vT[16 + r][j0 + kb16]);
      acc0 = wmma16(pa, vb0, acc0);
      acc1 = wmma16(pa, vb1, acc1);
    }

#pragma unroll
    for (int t = 0; t < 8; ++t) {
      const float  inv  = 1.0f / rsum[t];
      const size_t orow = head_base + (size_t)(m0 + t + mh) * D_;
      att[orow + r]      = (_Float16)(acc0[t] * inv);
      att[orow + 16 + r] = (_Float16)(acc1[t] * inv);
    }
  }
}

// ---- [3] output projection: out = att @ Wo^T -------------------------------
__global__ __launch_bounds__(256) void out_proj(
    const _Float16* __restrict__ att, const float* __restrict__ Wo,
    float* __restrict__ out) {
  const int lane = threadIdx.x & 31;
  const int wid  = blockIdx.x * 8 + (threadIdx.x >> 5);  // 0..1023
  const int m0   = wid * 16;
  const int r    = lane & 15;
  const int kb8  = (lane >> 4) << 3;
  const int kb16 = (lane >> 4) << 4;
  const int mh   = kb8;

  v16h a[8];
#pragma unroll
  for (int kk = 0; kk < 8; ++kk)
    a[kk] = afrag_f16(att + (size_t)(m0 + r) * D_ + kk * 32 + kb16);

  for (int nt = 0; nt < 16; ++nt) {
    v8f acc = {0.f, 0.f, 0.f, 0.f, 0.f, 0.f, 0.f, 0.f};
#pragma unroll
    for (int kk = 0; kk < 8; ++kk) {
      v16h b = bfrag_f32(Wo + (size_t)(nt * 16 + r) * D_ + kk * 32 + kb16);
      acc = wmma16(a[kk], b, acc);
    }
#pragma unroll
    for (int rr = 0; rr < 8; ++rr)
      out[(size_t)(m0 + rr + mh) * D_ + nt * 16 + r] = acc[rr];
  }
}

extern "C" void kernel_launch(void* const* d_in, const int* in_sizes, int n_in,
                              void* d_out, int out_size, void* d_ws, size_t ws_size,
                              hipStream_t stream) {
  (void)in_sizes; (void)n_in; (void)out_size; (void)ws_size;
  const float*         h    = (const float*)d_in[0];
  const float*         edge = (const float*)d_in[1];
  const unsigned char* adj  = (const unsigned char*)d_in[2];  // jnp bool = 1 byte
  const float*         Wq   = (const float*)d_in[3];
  const float*         Wk   = (const float*)d_in[4];
  const float*         Wv   = (const float*)d_in[5];
  const float*         We   = (const float*)d_in[6];          // [H,1]
  const float*         Wo   = (const float*)d_in[7];
  float*               out  = (float*)d_out;

  const size_t rows = (size_t)B_ * N_;   // 16384
  _Float16* qw = (_Float16*)d_ws;        // 8 MB each, 32 MB total
  _Float16* kw = qw + rows * D_;
  _Float16* vw = kw + rows * D_;
  _Float16* aw = vw + rows * D_;

  qkv_proj<<<dim3(384), dim3(256), 0, stream>>>(h, Wq, Wk, Wv, qw, kw, vw);
  edge_attn<<<dim3(B_ * H_), dim3(256), 0, stream>>>(qw, kw, vw, edge, adj, We, aw);
  out_proj<<<dim3(128), dim3(256), 0, stream>>>(aw, Wo, out);
}
